// Head_15023795601958
// MI455X (gfx1250) — matrix operand order repaired and checked by reading, not compiled, not run on hardware
//
#include <hip/hip_runtime.h>

typedef __attribute__((ext_vector_type(16))) _Float16 v16h;
typedef __attribute__((ext_vector_type(8)))  _Float16 v8h;
typedef __attribute__((ext_vector_type(8)))  float    v8f;
typedef __attribute__((ext_vector_type(4)))  float    v4f;

#define T_SEQ   8192
#define EMB     1024
#define HS      64
#define LOG2E_F 1.4426950408889634f
#define NEG_BIG (-2.0e9f)   // masked score in log2 domain; exp2 -> 0

static __device__ __forceinline__ v8f wmma_f16(v16h a, v16h b, v8f c) {
  // D = A(16x32 f16) * B(32x16 f16) + C(16x16 f32)
  return __builtin_amdgcn_wmma_f32_16x16x32_f16(false, a, false, b, (short)0, c,
                                                false, false);
}

// ---------------------------------------------------------------------------
// Kernel 1: Q/K/V projections  Y[t,h] = sum_e X[t,e] * W[h,e], cast to f16.
//   blockIdx.y selects matrix (0=Q row-major, 1=K row-major, 2=V transposed).
//   4 waves; wave w owns output columns [16w,16w+16). 32 chained WMMAs over E.
//   X tiles are re-read by the 4 waves but the whole input set (~96MB) fits in
//   the 192MB L2, so HBM traffic stays at the compulsory ~96MB.
// ---------------------------------------------------------------------------
__global__ __launch_bounds__(128) void qkv_proj(
    const float* __restrict__ Xq, const float* __restrict__ Xk,
    const float* __restrict__ Xv,
    const float* __restrict__ Wq, const float* __restrict__ Wk,
    const float* __restrict__ Wv,
    _Float16* __restrict__ Qh, _Float16* __restrict__ Kh,
    _Float16* __restrict__ Vt)
{
  const int which = blockIdx.y;
  const float* X = (which == 0) ? Xq : (which == 1) ? Xk : Xv;
  const float* W = (which == 0) ? Wq : (which == 1) ? Wk : Wv;

  const int lane  = threadIdx.x & 31;
  const int wave  = threadIdx.x >> 5;
  const int l15   = lane & 15;
  const int hi    = (lane >> 4) & 1;       // 16-lane half
  const int abase = hi * 8;                // A-layout K base for this half
  const int bko   = hi * 16;               // B-layout K base for this half
  const int t0    = blockIdx.x * 16;
  const int h0    = wave * 16;

  const float* xrow = X + (size_t)(t0 + l15) * EMB;  // A: row t0+l15
  const float* wrow = W + (size_t)(h0 + l15) * EMB;  // B: column h0+l15

  v8f acc = {};
  for (int kb = 0; kb < EMB; kb += 32) {
    v16h a, b;
    {
      // A 16x32 f16 layout: this half holds K = kb+bko+{0..15}
      v4f f0 = *(const v4f*)(xrow + kb + bko);
      v4f f1 = *(const v4f*)(xrow + kb + bko + 4);
      v4f f2 = *(const v4f*)(xrow + kb + bko + 8);
      v4f f3 = *(const v4f*)(xrow + kb + bko + 12);
#pragma unroll
      for (int e = 0; e < 4; ++e) {
        a[e]      = (_Float16)f0[e];
        a[4 + e]  = (_Float16)f1[e];
        a[8 + e]  = (_Float16)f2[e];
        a[12 + e] = (_Float16)f3[e];
      }
    }
    {
      // B 32x16 f16 layout: lane n holds column n, K = kb+bko+{0..15}
      v4f g0 = *(const v4f*)(wrow + kb + bko);
      v4f g1 = *(const v4f*)(wrow + kb + bko + 4);
      v4f g2 = *(const v4f*)(wrow + kb + bko + 8);
      v4f g3 = *(const v4f*)(wrow + kb + bko + 12);
#pragma unroll
      for (int e = 0; e < 4; ++e) {
        b[e]      = (_Float16)g0[e];
        b[4 + e]  = (_Float16)g1[e];
        b[8 + e]  = (_Float16)g2[e];
        b[12 + e] = (_Float16)g3[e];
      }
    }
    acc = wmma_f16(a, b, acc);
  }

  // C/D layout: VGPR r, lane -> row = t0 + hi*8 + r, col = h0 + l15
  const int colg  = h0 + l15;
  const int rbase = t0 + hi * 8;
  if (which == 2) {
    // V stored transposed: Vt[h][t]  (so P@V B-tiles are contiguous loads)
#pragma unroll
    for (int r = 0; r < 8; ++r)
      Vt[(size_t)colg * T_SEQ + (rbase + r)] = (_Float16)acc[r];
  } else {
    _Float16* O = (which == 0) ? Qh : Kh;
#pragma unroll
    for (int r = 0; r < 8; ++r)
      O[(size_t)(rbase + r) * HS + colg] = (_Float16)acc[r];
  }
}

// ---------------------------------------------------------------------------
// Kernel 2: fused causal flash attention, one wave per workgroup (512 groups
//   -> up to 32 single-wave groups per WGP for load balancing the triangle).
//   Per 64-key block: 8 WMMA (Q*K^T) -> log2-domain online softmax with
//   shfl_xor row-max only -> P relayout D->A via LDS -> 8 WMMA (P*V) +
//   2 WMMA (P*ones) that accumulate the softmax normalizer (no row-sum
//   shuffle reduction needed).
// ---------------------------------------------------------------------------
__global__ __launch_bounds__(32) void flash_attn(
    const _Float16* __restrict__ Qh, const _Float16* __restrict__ Kh,
    const _Float16* __restrict__ Vt, float* __restrict__ Out)
{
  __shared__ __align__(32) _Float16 pbuf[16 * 64];  // P tile, 2KB

  const int lane  = threadIdx.x;        // 0..31, single wave
  const int l15   = lane & 15;
  const int hi    = (lane >> 4) & 1;
  const int abase = hi * 8;
  const int t0    = blockIdx.x * 16;    // 16 query rows per wave
  const int qrow_base = t0 + hi * 8;

  // Load Q rows as two 16x32 A-tiles (head K-blocks 0 and 32)
  const _Float16* qrow = Qh + (size_t)(t0 + l15) * HS;
  v16h aq[2];
#pragma unroll
  for (int kt = 0; kt < 2; ++kt) {
    v8h lo = *(const v8h*)(qrow + kt * 32 + hi * 16);
    v8h hc = *(const v8h*)(qrow + kt * 32 + hi * 16 + 8);
#pragma unroll
    for (int e = 0; e < 8; ++e) { aq[kt][e] = lo[e]; aq[kt][8 + e] = hc[e]; }
  }

  v16h bones;           // all-ones B tile: P @ ones accumulates row sums
#pragma unroll
  for (int e = 0; e < 16; ++e) bones[e] = (_Float16)1.0f;

  v8f o[5] = {};        // o[0..3]: output head tiles, o[4]: softmax normalizer
  float m[8];
#pragma unroll
  for (int r = 0; r < 8; ++r) m[r] = -1e30f;

  const float k2 = 0.125f * LOG2E_F;    // scale * log2(e): softmax in 2^x domain
  const int nblk = (t0 + 16 + 63) >> 6; // causal: keys <= t0+15, 64-key blocks

  for (int blk = 0; blk < nblk; ++blk) {
    const int s0 = blk << 6;

    if (blk + 1 < nblk) {  // pull next K/V block toward the WGP
      __builtin_prefetch(Kh + (size_t)(s0 + 64 + lane) * HS, 0, 0);
      __builtin_prefetch(Kh + (size_t)(s0 + 96 + lane) * HS, 0, 0);
      __builtin_prefetch(Vt + (size_t)lane * T_SEQ + s0 + 64, 0, 0);
      __builtin_prefetch(Vt + (size_t)(32 + lane) * T_SEQ + s0 + 64, 0, 0);
    }

    // ---- S = Q * K^T : four 16-key column tiles ----
    v8f st[4] = {};
#pragma unroll
    for (int ct = 0; ct < 4; ++ct) {
      const _Float16* kp = Kh + (size_t)(s0 + ct * 16 + l15) * HS + hi * 16;
      st[ct] = wmma_f16(aq[0], *(const v16h*)(kp), st[ct]);
      st[ct] = wmma_f16(aq[1], *(const v16h*)(kp + 32), st[ct]);
    }

    // ---- scale into log2 domain (+ causal mask near the diagonal) ----
    float pmax[8];
    if (s0 + 63 > t0) {              // only diagonal-adjacent blocks mask
#pragma unroll
      for (int r = 0; r < 8; ++r) {
        const int qr = qrow_base + r;
        float mx = NEG_BIG;
#pragma unroll
        for (int ct = 0; ct < 4; ++ct) {
          float v = st[ct][r] * k2;
          if (s0 + ct * 16 + l15 > qr) v = NEG_BIG;
          st[ct][r] = v;
          mx = fmaxf(mx, v);
        }
        pmax[r] = mx;
      }
    } else {
#pragma unroll
      for (int r = 0; r < 8; ++r) {
        float mx = NEG_BIG;
#pragma unroll
        for (int ct = 0; ct < 4; ++ct) {
          float v = st[ct][r] * k2;
          st[ct][r] = v;
          mx = fmaxf(mx, v);
        }
        pmax[r] = mx;
      }
    }
    // row max across the 16 lanes of each half
#pragma unroll
    for (int xm = 1; xm < 16; xm <<= 1)
#pragma unroll
      for (int r = 0; r < 8; ++r)
        pmax[r] = fmaxf(pmax[r], __shfl_xor(pmax[r], xm, 32));

    // ---- online rescale + exponentiate ----
#pragma unroll
    for (int r = 0; r < 8; ++r) {
      const float mn    = fmaxf(m[r], pmax[r]);
      const float alpha = __builtin_amdgcn_exp2f(m[r] - mn);
      m[r] = mn;
#pragma unroll
      for (int nt = 0; nt < 5; ++nt) o[nt][r] *= alpha;
#pragma unroll
      for (int ct = 0; ct < 4; ++ct)
        st[ct][r] = __builtin_amdgcn_exp2f(st[ct][r] - mn);
    }

    // ---- D->A relayout of P through LDS (16x64 f16 row-major) ----
    const int prow = hi * 8;
#pragma unroll
    for (int r = 0; r < 8; ++r)
#pragma unroll
      for (int ct = 0; ct < 4; ++ct)
        pbuf[(prow + r) * 64 + ct * 16 + l15] = (_Float16)st[ct][r];
    // same-wave DS ops are in-order; drain before cross-lane readback
    __asm__ volatile("s_wait_dscnt 0" ::: "memory");
    v16h ap[2];
#pragma unroll
    for (int kt = 0; kt < 2; ++kt) {
      v8h lo = *(const v8h*)(pbuf + l15 * 64 + kt * 32 + hi * 16);
      v8h hc = *(const v8h*)(pbuf + l15 * 64 + kt * 32 + hi * 16 + 8);
#pragma unroll
      for (int e = 0; e < 8; ++e) { ap[kt][e] = lo[e]; ap[kt][8 + e] = hc[e]; }
    }

    // ---- O += P*V  and  normalizer += P*ones ----
#pragma unroll
    for (int nt = 0; nt < 4; ++nt) {
      const _Float16* vb = Vt + (size_t)(nt * 16 + l15) * T_SEQ + s0 + hi * 16;
      o[nt] = wmma_f16(ap[0], *(const v16h*)(vb), o[nt]);
      o[nt] = wmma_f16(ap[1], *(const v16h*)(vb + 32), o[nt]);
    }
    o[4] = wmma_f16(ap[0], bones, o[4]);
    o[4] = wmma_f16(ap[1], bones, o[4]);
  }

  // ---- normalize and store (o[4] columns all hold the row sum) ----
#pragma unroll
  for (int r = 0; r < 8; ++r) {
    const float inv = 1.0f / o[4][r];
    const size_t rowoff = (size_t)(qrow_base + r) * HS;
    Out[rowoff + 0  + l15] = o[0][r] * inv;
    Out[rowoff + 16 + l15] = o[1][r] * inv;
    Out[rowoff + 32 + l15] = o[2][r] * inv;
    Out[rowoff + 48 + l15] = o[3][r] * inv;
  }
}

// ---------------------------------------------------------------------------
extern "C" void kernel_launch(void* const* d_in, const int* in_sizes, int n_in,
                              void* d_out, int out_size, void* d_ws, size_t ws_size,
                              hipStream_t stream) {
  (void)in_sizes; (void)n_in; (void)out_size; (void)ws_size;
  const float* Xq = (const float*)d_in[0];
  const float* Xk = (const float*)d_in[1];
  const float* Xv = (const float*)d_in[2];
  // d_in[3] = boolean causal mask: 64 MB we never read (mask from indices)
  const float* Wq = (const float*)d_in[4];
  const float* Wk = (const float*)d_in[5];
  const float* Wv = (const float*)d_in[6];

  _Float16* Qh = (_Float16*)d_ws;                       // [T, 64] f16
  _Float16* Kh = Qh + (size_t)T_SEQ * HS;               // [T, 64] f16
  _Float16* Vt = Kh + (size_t)T_SEQ * HS;               // [64, T] f16 (transposed)

  qkv_proj<<<dim3(T_SEQ / 16, 3), 128, 0, stream>>>(Xq, Xk, Xv, Wq, Wk, Wv,
                                                    Qh, Kh, Vt);
  flash_attn<<<dim3(T_SEQ / 16), 32, 0, stream>>>(Qh, Kh, Vt, (float*)d_out);
}
